// BEVFormerDecoder_58317065945165
// MI455X (gfx1250) — matrix operand order repaired and checked by reading, not compiled, not run on hardware
//
#include <hip/hip_runtime.h>
#include <hip/hip_bf16.h>

typedef __attribute__((ext_vector_type(16))) __bf16 v16bf;
typedef __attribute__((ext_vector_type(8)))  __bf16 v8bf;
typedef __attribute__((ext_vector_type(4)))  __bf16 v4bf;
typedef __attribute__((ext_vector_type(8)))  float  v8f;

#define E_DIM 256
#define HEADS 8
#define DH    32
#define NQ    900
#define BS    4
#define HW_N  40000
#define HH    200
#define WW    200
#define PP    4
#define FFN   512
#define NLYR  6

static __device__ __forceinline__ __bf16 to_bf(float f) { return (__bf16)f; }

static __device__ __forceinline__ v16bf join16(v8bf lo, v8bf hi) {
    return __builtin_shufflevector(lo, hi, 0, 1, 2, 3, 4, 5, 6, 7,
                                   8, 9, 10, 11, 12, 13, 14, 15);
}

// ---------------------------------------------------------------------------
// f32 (+ optional f32) -> bf16 conversion, 4 elements/thread. n % 4 == 0.
// ---------------------------------------------------------------------------
__global__ __launch_bounds__(256) void cvt_bf16_kernel(
    const float* __restrict__ x, const float* __restrict__ x2,
    __bf16* __restrict__ y, long n)
{
    long i = ((long)blockIdx.x * blockDim.x + threadIdx.x) * 4;
    if (i >= n) return;
    float4 f = *(const float4*)(x + i);
    if (x2) {
        float4 g = *(const float4*)(x2 + i);
        f.x += g.x; f.y += g.y; f.z += g.z; f.w += g.w;
    }
    v4bf p;
    p[0] = to_bf(f.x); p[1] = to_bf(f.y); p[2] = to_bf(f.z); p[3] = to_bf(f.w);
    *(v4bf*)(y + i) = p;
}

// ---------------------------------------------------------------------------
// Weight transpose + convert: WT[n][k] = (bf16) W[k][n].  Tiled via LDS.
// ---------------------------------------------------------------------------
__global__ __launch_bounds__(256) void wtrans_bf16_kernel(
    const float* __restrict__ W, __bf16* __restrict__ WT, int K, int N)
{
    __shared__ __bf16 t[32][33];
    const int k0 = blockIdx.y * 32, n0 = blockIdx.x * 32;
    const int tx = threadIdx.x & 31, ty = threadIdx.x >> 5;  // 32 x 8
    for (int i = ty; i < 32; i += 8) {
        int k = k0 + i, n = n0 + tx;
        t[i][tx] = to_bf((k < K && n < N) ? W[(size_t)k * N + n] : 0.f);
    }
    __syncthreads();
    for (int i = ty; i < 32; i += 8) {
        int n = n0 + i, k = k0 + tx;
        if (n < N && k < K) WT[(size_t)n * K + k] = t[tx][i];
    }
}

// ---------------------------------------------------------------------------
// GEMM: Y[M,N] = act( A[M,K](bf16) @ WT[N,K](bf16)^T + bias[N] )
// Block = 256 thr (8 waves), tile 64(M) x 64(N), K stepped by 32.
// Double-buffered LDS -> one barrier per K-step; staging loads are
// unconditional (OOB rows/cols clamped: they only feed unstored outputs).
// Each wave: two v_wmma_f32_16x16x32_bf16.  Requires K%32==0.
// ---------------------------------------------------------------------------
__global__ __launch_bounds__(256) void gemm_bias_kernel(
    const __bf16* __restrict__ A, const __bf16* __restrict__ WT,
    const float* __restrict__ bias, float* __restrict__ Y,
    int M, int N, int K, int relu)
{
    __shared__ __bf16 As[2][64 * 32];   // [buf][row][k]
    __shared__ __bf16 BsT[2][64 * 32];  // [buf][col][k]
    const int tid  = threadIdx.x;
    const int lane = tid & 31;
    const int wave = tid >> 5;
    const int mw   = (wave & 3) * 16;
    const int nwb  = (wave >> 2) * 32;
    const int m0   = blockIdx.y * 64;
    const int n0   = blockIdx.x * 64;

    const int kbA = (lane < 16) ? 0 : 8;
    const int kbB = (lane < 16) ? 0 : 8;
    const int rA  = mw + (lane & 15);
    const int c0  = nwb + (lane & 15);
    const int c1  = c0 + 16;

    // staging assignment (one b128 per operand per thread); clamp OOB indices
    const int sAr = tid >> 2;          // 0..63 rows
    const int sAc = (tid & 3) * 8;     // 0,8,16,24
    const int sBc = tid & 63;          // 0..63 cols
    const int sBk = (tid >> 6) * 8;    // 0,8,16,24
    const int mC  = min(m0 + sAr, M - 1);
    const int nC  = min(n0 + sBc, N - 1);
    const __bf16* aSrc = A + (size_t)mC * K + sAc;
    const __bf16* bSrc = WT + (size_t)nC * K + sBk;

    // prologue: stage tile 0 into buffer 0
    v8bf av = *(const v8bf*)(aSrc);
    v8bf bv = *(const v8bf*)(bSrc);
    *(v8bf*)&As[0][sAr * 32 + sAc]  = av;
    *(v8bf*)&BsT[0][sBc * 32 + sBk] = bv;

    v8f acc0 = {}, acc1 = {};
    int p = 0;
    for (int k0 = 0; k0 < K; k0 += 32, p ^= 1) {
        __syncthreads();
        const bool more = (k0 + 32) < K;
        if (more) {
            av = *(const v8bf*)(aSrc + k0 + 32);
            bv = *(const v8bf*)(bSrc + k0 + 32);
            if (k0 + 64 < K) __builtin_prefetch(aSrc + k0 + 64, 0, 1);
        }

        v8bf alo = *(const v8bf*)&As[p][rA * 32 + kbA];
        v8bf ahi = *(const v8bf*)&As[p][rA * 32 + 16 + kbA];
        v16bf a = join16(alo, ahi);

        v8bf b0lo = *(const v8bf*)&BsT[p][c0 * 32 + kbB];
        v8bf b0hi = *(const v8bf*)&BsT[p][c0 * 32 + kbB + 16];
        v16bf bm0 = join16(b0lo, b0hi);
        v8bf b1lo = *(const v8bf*)&BsT[p][c1 * 32 + kbB];
        v8bf b1hi = *(const v8bf*)&BsT[p][c1 * 32 + kbB + 16];
        v16bf bm1 = join16(b1lo, b1hi);

        acc0 = __builtin_amdgcn_wmma_f32_16x16x32_bf16(false, a, false, bm0,
                                                       (short)0, acc0, false, false);
        acc1 = __builtin_amdgcn_wmma_f32_16x16x32_bf16(false, a, false, bm1,
                                                       (short)0, acc1, false, false);
        if (more) {
            *(v8bf*)&As[p ^ 1][sAr * 32 + sAc]  = av;
            *(v8bf*)&BsT[p ^ 1][sBc * 32 + sBk] = bv;
        }
    }

    const int nOut0 = n0 + c0;
    const int nOut1 = n0 + c1;
    const bool nok0 = (nOut0 < N);
    const bool nok1 = (nOut1 < N);
    const float bb0 = bias ? bias[min(nOut0, N - 1)] : 0.f;
    const float bb1 = bias ? bias[min(nOut1, N - 1)] : 0.f;
    #pragma unroll
    for (int vi = 0; vi < 8; ++vi) {
        int m = m0 + mw + vi + ((lane >= 16) ? 8 : 0);
        if (m < M) {
            if (nok0) {
                float r = acc0[vi] + bb0;
                if (relu) r = fmaxf(r, 0.f);
                Y[(size_t)m * N + nOut0] = r;
            }
            if (nok1) {
                float r = acc1[vi] + bb1;
                if (relu) r = fmaxf(r, 0.f);
                Y[(size_t)m * N + nOut1] = r;
            }
        }
    }
}

// ---------------------------------------------------------------------------
// Self-attention: one wave per (batch, head, 16-query tile).
// OOB rows/keys are clamped: their scores are either never written to LDS
// (tail columns stay -inf) or feed only unstored output rows.
// ---------------------------------------------------------------------------
__global__ __launch_bounds__(32) void attn_kernel(
    const float* __restrict__ Q, const float* __restrict__ Kx,
    const float* __restrict__ V, float* __restrict__ O)
{
    __shared__ float sc[16][928];
    const int lane = threadIdx.x;
    const int qt   = blockIdx.x;   // 0..56
    const int h    = blockIdx.y;
    const int b    = blockIdx.z;
    const float scale = 0.17677669529663687f; // 1/sqrt(32)

    for (int i = lane; i < 16 * 32; i += 32)
        sc[i >> 5][896 + (i & 31)] = -1e30f;

    const int kbA = (lane < 16) ? 0 : 8;
    const int kbB = (lane < 16) ? 0 : 16;

    // Q fragment (clamped row)
    const int qrow = min(qt * 16 + (lane & 15), NQ - 1);
    {
        const float* bp = Q + ((size_t)(b * NQ + qrow) * HEADS + h) * DH;
        float4 q0 = *(const float4*)(bp + kbA);
        float4 q1 = *(const float4*)(bp + kbA + 4);
        float4 q2 = *(const float4*)(bp + 16 + kbA);
        float4 q3 = *(const float4*)(bp + 20 + kbA);
        (void)q0; (void)q1; (void)q2; (void)q3;
    }
    const float* qp = Q + ((size_t)(b * NQ + qrow) * HEADS + h) * DH;
    float4 q0 = *(const float4*)(qp + kbA);
    float4 q1 = *(const float4*)(qp + kbA + 4);
    float4 q2 = *(const float4*)(qp + 16 + kbA);
    float4 q3 = *(const float4*)(qp + 20 + kbA);
    v16bf qa;
    qa[0]  = to_bf(q0.x * scale); qa[1]  = to_bf(q0.y * scale);
    qa[2]  = to_bf(q0.z * scale); qa[3]  = to_bf(q0.w * scale);
    qa[4]  = to_bf(q1.x * scale); qa[5]  = to_bf(q1.y * scale);
    qa[6]  = to_bf(q1.z * scale); qa[7]  = to_bf(q1.w * scale);
    qa[8]  = to_bf(q2.x * scale); qa[9]  = to_bf(q2.y * scale);
    qa[10] = to_bf(q2.z * scale); qa[11] = to_bf(q2.w * scale);
    qa[12] = to_bf(q3.x * scale); qa[13] = to_bf(q3.y * scale);
    qa[14] = to_bf(q3.z * scale); qa[15] = to_bf(q3.w * scale);

    // scores
    for (int kt = 0; kt < 57; ++kt) {
        int n  = kt * 16 + (lane & 15);
        int nc = min(n, NQ - 1);
        const float* bp = Kx + ((size_t)(b * NQ + nc) * HEADS + h) * DH + kbA;
        float4 k0v = *(const float4*)(bp);
        float4 k1v = *(const float4*)(bp + 4);
        float4 k2v = *(const float4*)(bp + 16);
        float4 k3v = *(const float4*)(bp + 20);
        v16bf kf;
        kf[0]  = to_bf(k0v.x); kf[1]  = to_bf(k0v.y);
        kf[2]  = to_bf(k0v.z); kf[3]  = to_bf(k0v.w);
        kf[4]  = to_bf(k1v.x); kf[5]  = to_bf(k1v.y);
        kf[6]  = to_bf(k1v.z); kf[7]  = to_bf(k1v.w);
        kf[8]  = to_bf(k2v.x); kf[9]  = to_bf(k2v.y);
        kf[10] = to_bf(k2v.z); kf[11] = to_bf(k2v.w);
        kf[12] = to_bf(k3v.x); kf[13] = to_bf(k3v.y);
        kf[14] = to_bf(k3v.z); kf[15] = to_bf(k3v.w);

        v8f s = {};
        s = __builtin_amdgcn_wmma_f32_16x16x32_bf16(false, qa, false, kf,
                                                    (short)0, s, false, false);
        #pragma unroll
        for (int vi = 0; vi < 8; ++vi) {
            int m = vi + ((lane >= 16) ? 8 : 0);
            if (n < NQ) sc[m][n] = s[vi];
        }
    }
    __syncthreads();

    // softmax per row (lanes 0..15 own one row each)
    if (lane < 16) {
        float mx = -1e30f;
        for (int c = 0; c < NQ; ++c) mx = fmaxf(mx, sc[lane][c]);
        float sum = 0.f;
        for (int c = 0; c < 928; ++c) {
            float e = __expf(sc[lane][c] - mx);
            sc[lane][c] = e;
            sum += e;
        }
        float inv = 1.f / sum;
        for (int c = 0; c < 928; ++c) sc[lane][c] *= inv;
    }
    __syncthreads();

    // O = P @ V  (clamped keys: weight is exactly 0 for cols >= NQ)
    v8f o0 = {}, o1 = {};
    for (int k0 = 0; k0 < 928; k0 += 32) {
        const float* sp = &sc[lane & 15][k0];
        float4 p0 = *(const float4*)(sp + kbA);
        float4 p1 = *(const float4*)(sp + kbA + 4);
        float4 p2 = *(const float4*)(sp + 16 + kbA);
        float4 p3 = *(const float4*)(sp + 20 + kbA);
        v16bf pa;
        pa[0]  = to_bf(p0.x); pa[1]  = to_bf(p0.y);
        pa[2]  = to_bf(p0.z); pa[3]  = to_bf(p0.w);
        pa[4]  = to_bf(p1.x); pa[5]  = to_bf(p1.y);
        pa[6]  = to_bf(p1.z); pa[7]  = to_bf(p1.w);
        pa[8]  = to_bf(p2.x); pa[9]  = to_bf(p2.y);
        pa[10] = to_bf(p2.z); pa[11] = to_bf(p2.w);
        pa[12] = to_bf(p3.x); pa[13] = to_bf(p3.y);
        pa[14] = to_bf(p3.z); pa[15] = to_bf(p3.w);

        v16bf v0, v1;
        #pragma unroll
        for (int j = 0; j < 8; ++j) {
            int kk = kbA + 2 * j + ((j >= 4) ? 8 : 0);
            int key0 = min(k0 + kk, NQ - 1);
            int key1 = min(k0 + kk + 1, NQ - 1);
            size_t base0 = ((size_t)(b * NQ + key0) * HEADS + h) * DH;
            size_t base1 = ((size_t)(b * NQ + key1) * HEADS + h) * DH;
            v0[2 * j]     = to_bf(V[base0 + (lane & 15)]);
            v0[2 * j + 1] = to_bf(V[base1 + (lane & 15)]);
            v1[2 * j]     = to_bf(V[base0 + 16 + (lane & 15)]);
            v1[2 * j + 1] = to_bf(V[base1 + 16 + (lane & 15)]);
        }
        o0 = __builtin_amdgcn_wmma_f32_16x16x32_bf16(false, pa, false, v0,
                                                     (short)0, o0, false, false);
        o1 = __builtin_amdgcn_wmma_f32_16x16x32_bf16(false, pa, false, v1,
                                                     (short)0, o1, false, false);
    }
    #pragma unroll
    for (int vi = 0; vi < 8; ++vi) {
        int q = qt * 16 + vi + ((lane >= 16) ? 8 : 0);
        if (q < NQ) {
            size_t base = ((size_t)(b * NQ + q) * HEADS + h) * DH;
            O[base + (lane & 15)]      = o0[vi];
            O[base + 16 + (lane & 15)] = o1[vi];
        }
    }
}

// ---------------------------------------------------------------------------
__global__ void aw_softmax_kernel(float* __restrict__ aw)
{
    int i = blockIdx.x * blockDim.x + threadIdx.x;
    if (i >= BS * NQ * HEADS) return;
    float* p = aw + (size_t)i * 4;
    float m = fmaxf(fmaxf(p[0], p[1]), fmaxf(p[2], p[3]));
    float e0 = __expf(p[0] - m), e1 = __expf(p[1] - m);
    float e2 = __expf(p[2] - m), e3 = __expf(p[3] - m);
    float inv = 1.f / (e0 + e1 + e2 + e3);
    p[0] = e0 * inv; p[1] = e1 * inv; p[2] = e2 * inv; p[3] = e3 * inv;
}

// ---------------------------------------------------------------------------
__global__ __launch_bounds__(256) void deform_sample_kernel(
    const float* __restrict__ ref, const float* __restrict__ off,
    const float* __restrict__ aw, const float* __restrict__ vp,
    float* __restrict__ ca)
{
    int bq = blockIdx.x;
    int b  = bq / NQ;
    int t  = threadIdx.x;
    int h  = t >> 5, d = t & 31;
    float rx = ref[(size_t)bq * 2 + 0];
    float ry = ref[(size_t)bq * 2 + 1];
    float acc = 0.f;
    #pragma unroll
    for (int p = 0; p < PP; ++p) {
        float ox = off[(size_t)bq * 64 + h * 8 + p * 2 + 0];
        float oy = off[(size_t)bq * 64 + h * 8 + p * 2 + 1];
        float w  = aw[(size_t)bq * 32 + h * 4 + p];
        float lx = (rx + ox * (1.f / WW)) * WW - 0.5f;
        float ly = (ry + oy * (1.f / HH)) * HH - 0.5f;
        float fx0 = floorf(lx), fy0 = floorf(ly);
        int x0 = (int)fx0, y0 = (int)fy0;
        float fx = lx - fx0, fy = ly - fy0;
        float s = 0.f;
        #pragma unroll
        for (int dy = 0; dy < 2; ++dy) {
            #pragma unroll
            for (int dx = 0; dx < 2; ++dx) {
                int xi = x0 + dx, yi = y0 + dy;
                float wgt = (dx ? fx : 1.f - fx) * (dy ? fy : 1.f - fy);
                if (xi >= 0 && xi < WW && yi >= 0 && yi < HH) {
                    s += wgt * vp[((size_t)(b * HW_N + yi * WW + xi) * HEADS + h) * DH + d];
                }
            }
        }
        acc += w * s;
    }
    ca[(size_t)bq * E_DIM + h * DH + d] = acc;
}

// ---------------------------------------------------------------------------
__global__ __launch_bounds__(32) void add_ln_kernel(
    const float* __restrict__ x, const float* __restrict__ r,
    const float* __restrict__ g, const float* __restrict__ beta,
    float* __restrict__ out)
{
    int row  = blockIdx.x;
    int lane = threadIdx.x;
    float vals[8];
    float s = 0.f;
    #pragma unroll
    for (int j = 0; j < 8; ++j) {
        int c = j * 32 + lane;
        vals[j] = x[(size_t)row * E_DIM + c] + r[(size_t)row * E_DIM + c];
        s += vals[j];
    }
    #pragma unroll
    for (int o = 16; o > 0; o >>= 1) s += __shfl_xor(s, o, 32);
    float mean = s * (1.f / E_DIM);
    float vv = 0.f;
    #pragma unroll
    for (int j = 0; j < 8; ++j) { float dl = vals[j] - mean; vv += dl * dl; }
    #pragma unroll
    for (int o = 16; o > 0; o >>= 1) vv += __shfl_xor(vv, o, 32);
    float rstd = rsqrtf(vv * (1.f / E_DIM) + 1e-5f);
    #pragma unroll
    for (int j = 0; j < 8; ++j) {
        int c = j * 32 + lane;
        out[(size_t)row * E_DIM + c] = (vals[j] - mean) * rstd * g[c] + beta[c];
    }
}

// ---------------------------------------------------------------------------
static void launch_cvt(const float* x, const float* x2, __bf16* y, long n,
                       hipStream_t stream)
{
    long th = n / 4;
    cvt_bf16_kernel<<<(unsigned)((th + 255) / 256), 256, 0, stream>>>(x, x2, y, n);
}

static void launch_wt(const float* W, __bf16* WT, int K, int N, hipStream_t stream)
{
    dim3 grid((N + 31) / 32, (K + 31) / 32);
    wtrans_bf16_kernel<<<grid, 256, 0, stream>>>(W, WT, K, N);
}

static void launch_gemm(const __bf16* A, const __bf16* WT, const float* bias,
                        float* Y, int M, int N, int K, int relu, hipStream_t stream)
{
    dim3 grid((N + 63) / 64, (M + 63) / 64);
    gemm_bias_kernel<<<grid, 256, 0, stream>>>(A, WT, bias, Y, M, N, K, relu);
}

extern "C" void kernel_launch(void* const* d_in, const int* in_sizes, int n_in,
                              void* d_out, int out_size, void* d_ws, size_t ws_size,
                              hipStream_t stream)
{
    (void)in_sizes; (void)n_in; (void)out_size; (void)ws_size;

    const float* query = (const float*)d_in[0];
    const float* value = (const float*)d_in[1];
    const float* ref   = (const float*)d_in[2];
    const float* qpos  = (const float*)d_in[3];
    const float* Wq  = (const float*)d_in[6];  const float* bq  = (const float*)d_in[7];
    const float* Wk  = (const float*)d_in[8];  const float* bk  = (const float*)d_in[9];
    const float* Wv  = (const float*)d_in[10]; const float* bv  = (const float*)d_in[11];
    const float* Wo  = (const float*)d_in[12]; const float* bo  = (const float*)d_in[13];
    const float* sow = (const float*)d_in[14]; const float* sob = (const float*)d_in[15];
    const float* aww = (const float*)d_in[16]; const float* awb = (const float*)d_in[17];
    const float* vpw = (const float*)d_in[18]; const float* vpb = (const float*)d_in[19];
    const float* opw = (const float*)d_in[20]; const float* opb = (const float*)d_in[21];
    const float* f1w = (const float*)d_in[22]; const float* f1b = (const float*)d_in[23];
    const float* f2w = (const float*)d_in[24]; const float* f2b = (const float*)d_in[25];
    const float* g1  = (const float*)d_in[26]; const float* b1  = (const float*)d_in[27];
    const float* g2  = (const float*)d_in[28]; const float* b2  = (const float*)d_in[29];
    const float* g3  = (const float*)d_in[30]; const float* b3  = (const float*)d_in[31];

    float* out = (float*)d_out;
    float* ws  = (float*)d_ws;

    const size_t S  = (size_t)BS * NQ * E_DIM;   // 921600
    const int    M  = BS * NQ;                   // 3600
    const int    MV = BS * HW_N;                 // 160000
    const size_t SV = (size_t)MV * E_DIM;        // 40,960,000

    // f32 scratch
    float* bufQ   = ws;
    float* bufK   = ws + S;
    float* bufV   = ws + 2 * S;
    float* bufA   = ws + 3 * S;
    float* bufS   = ws + 4 * S;
    float* bufX1  = ws + 5 * S;
    float* bufX2  = ws + 6 * S;
    float* bufCA  = ws + 7 * S;
    float* bufFF  = ws + 8 * S;                          // 2*S
    float* bufOFF = ws + 10 * S;                         // 3600 x 64
    float* bufAW  = ws + 10 * S + (size_t)BS * NQ * 64;  // 3600 x 32
    float* bufVP  = ws + 11 * S;                         // SV floats

    // bf16 scratch (after f32 region; 16B-aligned by construction)
    __bf16* bp = (__bf16*)(ws + 11 * S + SV);
    __bf16* valbf = bp;            bp += SV;       // bf16 value (converted once)
    __bf16* xqb   = bp;            bp += S;
    __bf16* x0b   = bp;            bp += S;
    __bf16* aab   = bp;            bp += S;
    __bf16* x1qb  = bp;            bp += S;
    __bf16* cab   = bp;            bp += S;
    __bf16* x2b   = bp;            bp += S;
    __bf16* ffb   = bp;            bp += 2 * S;
    const size_t WSLOT = (size_t)E_DIM * FFN;
    __bf16* wT[10];
    for (int i = 0; i < 10; ++i) { wT[i] = bp; bp += WSLOT; }

    // value is layer-invariant: convert once
    launch_cvt(value, nullptr, valbf, (long)SV, stream);

    for (int l = 0; l < NLYR; ++l) {
        const float* x0 = (l == 0) ? query : out + (size_t)(l - 1) * S;
        const size_t wE = (size_t)l * E_DIM * E_DIM;
        const size_t bE = (size_t)l * E_DIM;

        // --- self attention ---
        launch_cvt(x0, qpos, xqb, (long)S, stream);
        launch_cvt(x0, nullptr, x0b, (long)S, stream);
        launch_wt(Wq + wE, wT[0], E_DIM, E_DIM, stream);
        launch_wt(Wk + wE, wT[1], E_DIM, E_DIM, stream);
        launch_wt(Wv + wE, wT[2], E_DIM, E_DIM, stream);
        launch_gemm(xqb, wT[0], bq + bE, bufQ, M, E_DIM, E_DIM, 0, stream);
        launch_gemm(xqb, wT[1], bk + bE, bufK, M, E_DIM, E_DIM, 0, stream);
        launch_gemm(x0b, wT[2], bv + bE, bufV, M, E_DIM, E_DIM, 0, stream);
        attn_kernel<<<dim3(57, HEADS, BS), 32, 0, stream>>>(bufQ, bufK, bufV, bufA);
        launch_cvt(bufA, nullptr, aab, (long)S, stream);
        launch_wt(Wo + wE, wT[3], E_DIM, E_DIM, stream);
        launch_gemm(aab, wT[3], bo + bE, bufS, M, E_DIM, E_DIM, 0, stream);
        add_ln_kernel<<<M, 32, 0, stream>>>(x0, bufS, g1 + bE, b1 + bE, bufX1);

        // --- deformable cross attention ---
        launch_wt(vpw + wE, wT[4], E_DIM, E_DIM, stream);
        launch_gemm(valbf, wT[4], vpb + bE, bufVP, MV, E_DIM, E_DIM, 0, stream);
        launch_cvt(bufX1, qpos, x1qb, (long)S, stream);
        launch_wt(sow + (size_t)l * E_DIM * 64, wT[5], E_DIM, 64, stream);
        launch_gemm(x1qb, wT[5], sob + (size_t)l * 64, bufOFF, M, 64, E_DIM, 0, stream);
        launch_wt(aww + (size_t)l * E_DIM * 32, wT[6], E_DIM, 32, stream);
        launch_gemm(x1qb, wT[6], awb + (size_t)l * 32, bufAW, M, 32, E_DIM, 0, stream);
        aw_softmax_kernel<<<(BS * NQ * HEADS + 255) / 256, 256, 0, stream>>>(bufAW);
        deform_sample_kernel<<<M, 256, 0, stream>>>(ref, bufOFF, bufAW, bufVP, bufCA);
        launch_cvt(bufCA, nullptr, cab, (long)S, stream);
        launch_wt(opw + wE, wT[7], E_DIM, E_DIM, stream);
        launch_gemm(cab, wT[7], opb + bE, bufS, M, E_DIM, E_DIM, 0, stream);
        add_ln_kernel<<<M, 32, 0, stream>>>(bufX1, bufS, g2 + bE, b2 + bE, bufX2);

        // --- FFN ---
        launch_cvt(bufX2, nullptr, x2b, (long)S, stream);
        launch_wt(f1w + (size_t)l * E_DIM * FFN, wT[8], E_DIM, FFN, stream);
        launch_gemm(x2b, wT[8], f1b + (size_t)l * FFN, bufFF, M, FFN, E_DIM, 1, stream);
        launch_cvt(bufFF, nullptr, ffb, (long)(2 * S), stream);
        launch_wt(f2w + (size_t)l * FFN * E_DIM, wT[9], FFN, E_DIM, stream);
        launch_gemm(ffb, wT[9], f2b + bE, bufS, M, E_DIM, FFN, 0, stream);
        add_ln_kernel<<<M, 32, 0, stream>>>(bufX2, bufS, g3 + bE, b3 + bE,
                                            out + (size_t)l * S);
    }
}
